// EnsembleModel2_50044958933021
// MI455X (gfx1250) — matrix operand order repaired and checked by reading, not compiled, not run on hardware
//
#include <hip/hip_runtime.h>
#include <hip/hip_bf16.h>
#include <hip/hip_fp16.h>

#define NG 368
#define NM 64
#define NB 4096

typedef __attribute__((ext_vector_type(16))) _Float16 v16h;
typedef __attribute__((ext_vector_type(8)))  _Float16 v8h;
typedef __attribute__((ext_vector_type(8)))  float    v8f;
typedef __attribute__((ext_vector_type(4)))  float    v4f;

union V16H { v16h v; v8h h[2]; };

// One block: one group g, 512 batch rows (8 waves x 16-row tile x 4 iters).
// Per wave-tile: Y[16x64] = X[16x64] @ W[g]^T + b[g] via 8x v_wmma_f32_16x16x32_f16,
// then softmax over 64 cols fused with the resp·x reduction.
__global__ __launch_bounds__(256)
void ensemble_grouped_softmax_kernel(const float* __restrict__ x,
                                     const float* __restrict__ W,
                                     const float* __restrict__ bias,
                                     float* __restrict__ out) {
  __shared__ __align__(16) _Float16 sW[64][72];     // W[g] as f16, [m][n], padded pitch
  __shared__ float sB[64];                          // bias[g]
  __shared__ __align__(16) _Float16 sX[8][16][72];  // per-wave X tile, f16
  __shared__ __align__(16) float    sY[8][16][68];  // per-wave Y tile, f32

  const int tid  = threadIdx.x;
  const int w    = tid >> 5;
  const int lane = tid & 31;
  const int g    = blockIdx.y;
  const int bblk = blockIdx.x;       // 0..7
  const int lo   = lane & 15;
  const int hi   = lane >> 4;

  // ---- stage W[g] (64x64 f32 -> f16) and bias into LDS ----
  {
    const float* Wg = W + (size_t)g * NM * NM;
    const int m  = tid >> 2;          // 0..63
    const int c0 = (tid & 3) * 16;    // 0,16,32,48
#pragma unroll
    for (int i = 0; i < 4; ++i) {
      v4f v = *(const v4f*)(Wg + m * NM + c0 + i * 4);
      _Float16* dst = &sW[m][c0 + i * 4];
      dst[0] = (_Float16)v.x; dst[1] = (_Float16)v.y;
      dst[2] = (_Float16)v.z; dst[3] = (_Float16)v.w;
    }
    if (tid < 64) sB[tid] = bias[(size_t)g * NM + tid];
  }
  __syncthreads();

  // ---- hoist B = W[g]^T into registers: 4 N-tiles x 2 K-chunks ----
  // B 32x16 f16, contiguous k order used here (matches A): lanes 0-15 col j hold K=0..15, lanes 16-31 K=16..31,
  // i.e. 16 contiguous n-values of W[g][j][*].
  V16H Bt[8];
#pragma unroll
  for (int t = 0; t < 4; ++t) {
    const int j = t * 16 + lo;
#pragma unroll
    for (int kc = 0; kc < 2; ++kc) {
      const _Float16* p = &sW[j][kc * 32 + hi * 16];
      Bt[t * 2 + kc].h[0] = *(const v8h*)(p);
      Bt[t * 2 + kc].h[1] = *(const v8h*)(p + 8);
    }
  }
  float bc[4];
#pragma unroll
  for (int t = 0; t < 4; ++t) bc[t] = sB[t * 16 + lo];

  for (int it = 0; it < 4; ++it) {
    const int row0 = bblk * 512 + (it * 8 + w) * 16;  // first batch row of this tile

    // ---- stage X tile: 16 rows x 64 cols, f32 -> f16 in LDS ----
    {
      const int r  = lane >> 1;
      const int cb = (lane & 1) * 32;
      const float* xp = x + ((size_t)(row0 + r) * NG + g) * NM + cb;
#pragma unroll
      for (int i = 0; i < 8; ++i) {
        v4f v = *(const v4f*)(xp + i * 4);
        _Float16* dst = &sX[w][r][cb + i * 4];
        dst[0] = (_Float16)v.x; dst[1] = (_Float16)v.y;
        dst[2] = (_Float16)v.z; dst[3] = (_Float16)v.w;
      }
    }
    __syncthreads();

    // ---- GEMM: acc[t] starts at bias (bias depends only on column) ----
    v8f acc[4];
#pragma unroll
    for (int t = 0; t < 4; ++t)
#pragma unroll
      for (int v = 0; v < 8; ++v) acc[t][v] = bc[t];

#pragma unroll
    for (int kc = 0; kc < 2; ++kc) {
      // A 16x32 f16, contiguous k order used here (same as B): lane<16 row=lo K=0..15; lane>=16 K=16..31
      V16H a;
      const _Float16* ap = &sX[w][lo][kc * 32 + hi * 16];
      a.h[0] = *(const v8h*)(ap);
      a.h[1] = *(const v8h*)(ap + 8);
#pragma unroll
      for (int t = 0; t < 4; ++t) {
        acc[t] = __builtin_amdgcn_wmma_f32_16x16x32_f16(
            false, a.v, false, Bt[t * 2 + kc].v, (short)0, acc[t], false, false);
      }
    }

    // ---- spill Y tile to LDS (row = v + 8*hi, col = t*16 + lo) ----
#pragma unroll
    for (int t = 0; t < 4; ++t)
#pragma unroll
      for (int v = 0; v < 8; ++v)
        sY[w][v + 8 * hi][t * 16 + lo] = acc[t][v];
    __syncthreads();

    // ---- fused softmax + dot(resp, x): one lane per row ----
    if (lane < 16) {
      const float* yr = sY[w][lane];
      const _Float16* xr = sX[w][lane];
      float mx = -3.0e38f;
#pragma unroll
      for (int c4 = 0; c4 < 16; ++c4) {
        v4f yv = *(const v4f*)(yr + c4 * 4);
        mx = fmaxf(mx, fmaxf(fmaxf(yv.x, yv.y), fmaxf(yv.z, yv.w)));
      }
      float den = 0.f, num = 0.f;
#pragma unroll
      for (int c4 = 0; c4 < 16; ++c4) {
        v4f yv = *(const v4f*)(yr + c4 * 4);
        const _Float16* xp4 = xr + c4 * 4;
        float e0 = __expf(yv.x - mx), e1 = __expf(yv.y - mx);
        float e2 = __expf(yv.z - mx), e3 = __expf(yv.w - mx);
        den += (e0 + e1) + (e2 + e3);
        num += e0 * (float)xp4[0] + e1 * (float)xp4[1] +
               e2 * (float)xp4[2] + e3 * (float)xp4[3];
      }
      out[(size_t)(row0 + lane) * NG + g] = num / den;
    }
    __syncthreads();
  }
}

extern "C" void kernel_launch(void* const* d_in, const int* in_sizes, int n_in,
                              void* d_out, int out_size, void* d_ws, size_t ws_size,
                              hipStream_t stream) {
  const float* x = (const float*)d_in[0];
  const float* W = (const float*)d_in[1];
  const float* b = (const float*)d_in[2];
  float* out = (float*)d_out;
  dim3 grid(8, NG, 1);  // 8 batch-blocks (512 rows each) x 368 groups
  ensemble_grouped_softmax_kernel<<<grid, 256, 0, stream>>>(x, W, b, out);
}
